// VectorQuantizer_17162689315041
// MI455X (gfx1250) — matrix operand order repaired and checked by reading, not compiled, not run on hardware
//
#include <hip/hip_runtime.h>

typedef __attribute__((ext_vector_type(16))) _Float16 v16h;
typedef __attribute__((ext_vector_type(8)))  _Float16 v8h;
typedef __attribute__((ext_vector_type(8)))  float    v8f;

#define NUM_EMB   1024
#define DIM       128
#define N_VEC     32768      // 32 * 32 * 32
#define OUT_ELEMS 4194304    // 32 * 128 * 32 * 32

// workspace layout (bytes)
#define WS_EMBH_OFF  0                              // 1024*128 f16 = 256 KB
#define WS_ENORM_OFF (NUM_EMB * DIM * 2)            // 1024 f32   = 4 KB
#define WS_LATH_OFF  (WS_ENORM_OFF + NUM_EMB * 4)   // 32768*128 f16 = 8 MB

// ---------------------------------------------------------------------------
// Kernel 1: codebook f32 -> f16, exact f32 row norms, zero loss slot.
// grid = 1024 blocks (one code each), 128 threads.
// ---------------------------------------------------------------------------
__global__ void prep_embedding(const float* __restrict__ emb,
                               _Float16* __restrict__ emb_h,
                               float* __restrict__ enorm,
                               float* __restrict__ loss_slot) {
  int k = blockIdx.x;
  int d = threadIdx.x;
  float v = emb[k * DIM + d];
  emb_h[k * DIM + d] = (_Float16)v;
  float s = v * v;
#pragma unroll
  for (int off = 16; off > 0; off >>= 1) s += __shfl_xor(s, off);
  __shared__ float red[4];
  int wave = threadIdx.x >> 5, lane = threadIdx.x & 31;
  if (lane == 0) red[wave] = s;
  __syncthreads();
  if (threadIdx.x == 0) {
    enorm[k] = red[0] + red[1] + red[2] + red[3];
    if (k == 0) *loss_slot = 0.0f;
  }
}

// ---------------------------------------------------------------------------
// Kernel 2: transpose latents [B,D,H,W] f32 -> flat [N=B*H*W, D] f16.
// grid = B*H = 1024 blocks, 256 threads, LDS tile 128(d) x 32(w), padded.
// ---------------------------------------------------------------------------
__global__ void prep_latents(const float* __restrict__ lat,
                             _Float16* __restrict__ lat_h) {
  int bh = blockIdx.x;
  int b = bh >> 5, h = bh & 31;
  __shared__ float tile[128][33];
  int t = threadIdx.x;
  const float* src = lat + (size_t)b * 131072 + h * 32;  // + d*1024 + w
#pragma unroll
  for (int i = 0; i < 16; ++i) {
    int idx = t + i * 256;
    int d = idx >> 5, w = idx & 31;
    tile[d][w] = src[(size_t)d * 1024 + w];              // coalesced in w
  }
  __syncthreads();
  _Float16* dst = lat_h + (size_t)(b * 1024 + h * 32) * DIM;  // + w*128 + d
#pragma unroll
  for (int i = 0; i < 16; ++i) {
    int idx = t + i * 256;
    int d = idx & 127, w = idx >> 7;
    dst[(size_t)w * DIM + d] = (_Float16)tile[d][w];     // coalesced in d
  }
}

// ---------------------------------------------------------------------------
// Kernel 3: WMMA distance GEMM + argmin + gather + loss.
// grid = N/32 = 1024 blocks, 128 threads (4 waves).
// Each wave: rows [n0, n0+32) as two 16-row A tiles, codes [wave*256, +256).
// ---------------------------------------------------------------------------
__launch_bounds__(128, 1)
__global__ void vq_main(const _Float16* __restrict__ lat_h,
                        const _Float16* __restrict__ emb_h,
                        const float* __restrict__ enorm,
                        const float* __restrict__ emb,
                        const float* __restrict__ lat,
                        float* __restrict__ qout,
                        float* __restrict__ loss) {
  int tid  = threadIdx.x;
  int wave = tid >> 5, lane = tid & 31;
  int mrow = lane & 15, half = lane >> 4;
  int n0 = blockIdx.x * 32;

  // A fragments: 16-bit A 16x32 layout (ISA 7.12.2):
  // elements 0..7 = K(half*8 .. +8), elements 8..15 = K(16+half*8 .. +8)
  v16h a0[4], a1[4];
#pragma unroll
  for (int ks = 0; ks < 4; ++ks) {
    const _Float16* p0 = lat_h + (size_t)(n0 + mrow) * DIM + ks * 32 + half * 8;
    v8h lo = *(const v8h*)p0;
    v8h hi = *(const v8h*)(p0 + 16);
    a0[ks] = __builtin_shufflevector(lo, hi, 0,1,2,3,4,5,6,7,8,9,10,11,12,13,14,15);
    const _Float16* p1 = p0 + 16 * DIM;
    v8h lo1 = *(const v8h*)p1;
    v8h hi1 = *(const v8h*)(p1 + 16);
    a1[ks] = __builtin_shufflevector(lo1, hi1, 0,1,2,3,4,5,6,7,8,9,10,11,12,13,14,15);
  }

  float bv0[8], bv1[8];
  int   bi0[8], bi1[8];
#pragma unroll
  for (int r = 0; r < 8; ++r) {
    bv0[r] = 3.4e38f; bv1[r] = 3.4e38f; bi0[r] = 0; bi1[r] = 0;
  }

  for (int tt = 0; tt < 16; ++tt) {
    int c = (wave * 16 + tt) * 16 + mrow;    // code handled by this lane
    float en = enorm[c];
    v8f acc0 = {}; v8f acc1 = {};
#pragma unroll
    for (int ks = 0; ks < 4; ++ks) {
      // B 32x16 layout: elements 0..7 = K(half*8 .. +8), 8..15 = K(16+half*8 .. +8) of code row c
      const _Float16* pb = emb_h + (size_t)c * DIM + ks * 32 + half * 8;
      v8h blo = *(const v8h*)pb;
      v8h bhi = *(const v8h*)(pb + 16);
      v16h bf = __builtin_shufflevector(blo, bhi, 0,1,2,3,4,5,6,7,8,9,10,11,12,13,14,15);
      acc0 = __builtin_amdgcn_wmma_f32_16x16x32_f16(false, a0[ks], false, bf,
                                                    (short)0, acc0, false, false);
      acc1 = __builtin_amdgcn_wmma_f32_16x16x32_f16(false, a1[ks], false, bf,
                                                    (short)0, acc1, false, false);
    }
#pragma unroll
    for (int r = 0; r < 8; ++r) {
      float d0 = en - 2.0f * acc0[r];
      if (d0 < bv0[r]) { bv0[r] = d0; bi0[r] = c; }
      float d1 = en - 2.0f * acc1[r];
      if (d1 < bv1[r]) { bv1[r] = d1; bi1[r] = c; }
    }
  }

  __shared__ float s_val[4][2][16];
  __shared__ int   s_idx[4][2][16];
  __shared__ int   s_fin[32];
  __shared__ float s_red[4];

  // min across the 16 code-lanes of each half-wave (xor masks stay in-group)
#pragma unroll
  for (int r = 0; r < 8; ++r) {
    float v0 = bv0[r]; int i0 = bi0[r];
    float v1 = bv1[r]; int i1 = bi1[r];
#pragma unroll
    for (int off = 8; off > 0; off >>= 1) {
      float ov = __shfl_xor(v0, off); int oi = __shfl_xor(i0, off);
      if (ov < v0 || (ov == v0 && oi < i0)) { v0 = ov; i0 = oi; }
      ov = __shfl_xor(v1, off); oi = __shfl_xor(i1, off);
      if (ov < v1 || (ov == v1 && oi < i1)) { v1 = ov; i1 = oi; }
    }
    if (mrow == 0) {                      // lanes 0 and 16
      int m = half * 8 + r;               // row within 16-row tile
      s_val[wave][0][m] = v0; s_idx[wave][0][m] = i0;
      s_val[wave][1][m] = v1; s_idx[wave][1][m] = i1;
    }
  }
  __syncthreads();

  // reduce over the 4 waves (waves own increasing code ranges -> tie-break ok)
  if (tid < 32) {
    int t = tid >> 4, m = tid & 15;
    float bv = s_val[0][t][m]; int bi = s_idx[0][t][m];
#pragma unroll
    for (int w = 1; w < 4; ++w) {
      float v = s_val[w][t][m]; int i = s_idx[w][t][m];
      if (v < bv || (v == bv && i < bi)) { bv = v; bi = i; }
    }
    s_fin[tid] = bi;                      // row tid of this block
  }
  __syncthreads();

  // Gather + write q in [B,D,H,W] order + fused loss.
  // lanes cover 32 consecutive hw (coalesced stores); wave selects d-chunk.
  int n = n0 + lane;
  int bb = n >> 10, hw = n & 1023;
  const float* erow = emb + (size_t)s_fin[lane] * DIM;
  size_t base = (size_t)bb * 131072 + hw;
  float sum = 0.0f;
#pragma unroll 4
  for (int j = 0; j < 32; ++j) {
    int d = wave * 32 + j;
    size_t a = base + (size_t)d * 1024;
    float q = erow[d];
    float l = lat[a];
    float df = q - l;
    sum += df * df;
    qout[a] = l + df;                     // straight-through: lat + (q - lat)
  }
#pragma unroll
  for (int off = 16; off > 0; off >>= 1) sum += __shfl_xor(sum, off);
  if (lane == 0) s_red[wave] = sum;
  __syncthreads();
  if (tid == 0) {
    float tot = (s_red[0] + s_red[1] + s_red[2] + s_red[3]) *
                (1.25f / (float)OUT_ELEMS);   // (1 + BETA) * mean
    atomicAdd(loss, tot);
  }
}

// ---------------------------------------------------------------------------
extern "C" void kernel_launch(void* const* d_in, const int* in_sizes, int n_in,
                              void* d_out, int out_size, void* d_ws, size_t ws_size,
                              hipStream_t stream) {
  (void)in_sizes; (void)n_in; (void)out_size; (void)ws_size;
  const float* latents   = (const float*)d_in[0];
  const float* embedding = (const float*)d_in[1];
  float* out = (float*)d_out;
  char* ws = (char*)d_ws;
  _Float16* emb_h = (_Float16*)(ws + WS_EMBH_OFF);
  float*    enorm = (float*)(ws + WS_ENORM_OFF);
  _Float16* lat_h = (_Float16*)(ws + WS_LATH_OFF);
  float*    loss  = out + OUT_ELEMS;

  prep_embedding<<<NUM_EMB, 128, 0, stream>>>(embedding, emb_h, enorm, loss);
  prep_latents<<<1024, 256, 0, stream>>>(latents, lat_h);
  vq_main<<<1024, 128, 0, stream>>>(lat_h, emb_h, enorm, embedding, latents, out, loss);
}
